// MultiHeadAttention_61091614818698
// MI455X (gfx1250) — matrix operand order repaired and checked by reading, not compiled, not run on hardware
//
#include <hip/hip_runtime.h>
#include <hip/hip_bf16.h>
#include <stdint.h>

// ---------------------------------------------------------------------------
// MHA forward for MI455X (gfx1250, wave32).
// bf16 WMMA (v_wmma_f32_16x16x32_bf16) for every GEMM, f32 accumulate,
// f32 softmax. K/V chunks staged with GLOBAL_LOAD_ASYNC_TO_LDS_B128
// (ASYNCcnt). Workspace (needs 32 MB): Q | K | V^T | O, all bf16.
// ---------------------------------------------------------------------------

typedef __bf16 bf16;
typedef __attribute__((ext_vector_type(16))) __bf16 v16bf;
typedef __attribute__((ext_vector_type(8)))  float  v8f;

#define D_MODEL 1024
#define N_HEADS 16
#define HEAD    64
#define SEQ     2048
#define BATCH   2
#define TOKENS  (BATCH * SEQ)   // 4096

#if defined(__has_builtin)
#if __has_builtin(__builtin_amdgcn_cvt_pk_bf16_f32)
#define HAVE_PK_BF16 1
#endif
#endif

// f32 -> bf16, round-half-up (tie cases differ from RNE with prob 2^-16,
// far below bf16 quantization noise). Pair pack = 2 adds + 1 v_perm_b32.
__device__ __forceinline__ unsigned pk2(float lo, float hi) {
#if defined(HAVE_PK_BF16)
  auto r = __builtin_amdgcn_cvt_pk_bf16_f32(lo, hi);   // v_cvt_pk_bf16_f32
  return __builtin_bit_cast(unsigned, r);
#else
  unsigned ul = __builtin_bit_cast(unsigned, lo) + 0x8000u;
  unsigned uh = __builtin_bit_cast(unsigned, hi) + 0x8000u;
  // bytes {uh[3],uh[2],ul[3],ul[2]} -> {hi_bf16, lo_bf16}
  return __builtin_amdgcn_perm(uh, ul, 0x07060302u);
#endif
}
__device__ __forceinline__ bf16 f2bf(float f) {
  unsigned u = __builtin_bit_cast(unsigned, f) + 0x8000u;
  unsigned short h = (unsigned short)(u >> 16);
  return __builtin_bit_cast(bf16, h);
}

union FragAB { v16bf v; uint4 q[2]; };

#define WMMA_BF16(A, B, C) \
  __builtin_amdgcn_wmma_f32_16x16x32_bf16(false, (A), false, (B), (short)0, (C), false, false)

// Async copy 16 bytes global -> LDS (per lane), ASYNCcnt-tracked (CDNA5).
// INST_OFFSET is added to BOTH the LDS and global addresses (ISA 08 §4.4).
#define ASYNC_G2L_B128(ldsoff_u32, gaddr_u64, imm_off)                         \
  asm volatile("global_load_async_to_lds_b128 %0, %1, off offset:" #imm_off   \
               :: "v"(ldsoff_u32), "v"(gaddr_u64) : "memory")

__device__ __forceinline__ unsigned lds_off(const void* p) {
  return (unsigned)(uintptr_t)p;   // generic->LDS: hardware truncates to addr[31:0]
}

// ---------------------------------------------------------------------------
// Kernel 1: QKV projection.  grid = (Mtiles=32, heads=16, type=3), block=256.
// Out: Q,K row-major bf16 [b,h,s,64]; V transposed bf16 [b,h,64,s].
// ---------------------------------------------------------------------------
__global__ __launch_bounds__(256)
void qkv_proj_kernel(const float* __restrict__ x,
                     const float* __restrict__ Wq, const float* __restrict__ bq,
                     const float* __restrict__ Wk, const float* __restrict__ bk,
                     const float* __restrict__ Wv, const float* __restrict__ bv,
                     bf16* __restrict__ Qw, bf16* __restrict__ Kw,
                     bf16* __restrict__ Vtw)
{
  const int mt   = blockIdx.x;        // 128-token tile
  const int h    = blockIdx.y;
  const int ty   = blockIdx.z;        // 0=Q 1=K 2=V
  const int tid  = threadIdx.x;
  const int wave = tid >> 5, lane = tid & 31;

  const float* W  = (ty == 0 ? Wq : ty == 1 ? Wk : Wv) + (size_t)h * D_MODEL * HEAD;
  const float* bs = (ty == 0 ? bq : ty == 1 ? bk : bv) + h * HEAD;

  __shared__ __align__(16) bf16 a_s[128][40];   // 128 x 32 (pitch 40)
  __shared__ __align__(16) bf16 bT_s[64][40];   // n=64 x k=32 (B stored transposed)

  const int m0 = mt * 128;
  v8f acc[4] = {};

  for (int k0 = 0; k0 < D_MODEL; k0 += 32) {
    __syncthreads();
    { // stage A: x[m0+row][k0..k0+32) -> bf16
      int row = tid >> 1, c = (tid & 1) * 16;
      const float4* src = (const float4*)&x[(size_t)(m0 + row) * D_MODEL + k0 + c];
      float4 f0 = src[0], f1 = src[1], f2 = src[2], f3 = src[3];
      uint4 p0 = { pk2(f0.x,f0.y), pk2(f0.z,f0.w), pk2(f1.x,f1.y), pk2(f1.z,f1.w) };
      uint4 p1 = { pk2(f2.x,f2.y), pk2(f2.z,f2.w), pk2(f3.x,f3.y), pk2(f3.z,f3.w) };
      uint4* dst = (uint4*)&a_s[row][c];
      dst[0] = p0; dst[1] = p1;
      if (k0 + 32 < D_MODEL)
        __builtin_prefetch(&x[(size_t)(m0 + row) * D_MODEL + k0 + 32 + c], 0, 0);
    }
    { // stage B transposed: bT_s[n][k] = W[(k0+k)*64 + n]
      int k = tid >> 3, n = (tid & 7) * 8;
      const float4* src = (const float4*)&W[(size_t)(k0 + k) * HEAD + n];
      float4 f0 = src[0], f1 = src[1];
      float v[8] = { f0.x,f0.y,f0.z,f0.w, f1.x,f1.y,f1.z,f1.w };
      #pragma unroll
      for (int j = 0; j < 8; ++j) bT_s[n + j][k] = f2bf(v[j]);
    }
    __syncthreads();

    const int row  = lane & 15;
    const int klo  = (lane >> 4) * 8;    // A frag: K 0-7 / 8-15 split per half
    const int col  = lane & 15;
    const int koff = (lane >> 4) * 16;   // unused: B frags use klo, the same K split as A
    FragAB fa;
    fa.q[0] = *(const uint4*)&a_s[wave * 16 + row][klo];
    fa.q[1] = *(const uint4*)&a_s[wave * 16 + row][16 + klo];
    #pragma unroll
    for (int nt = 0; nt < 4; ++nt) {
      FragAB fb;
      fb.q[0] = *(const uint4*)&bT_s[nt * 16 + col][klo];
      fb.q[1] = *(const uint4*)&bT_s[nt * 16 + col][16 + klo];
      acc[nt] = WMMA_BF16(fa.v, fb.v, acc[nt]);
    }
  }

  // epilogue: + bias, store bf16 (C layout: lanes 0-15 rows 0-7, 16-31 rows 8-15)
  const int b     = m0 / SEQ;
  const int s0    = m0 % SEQ;
  const int mr_lo = (lane < 16) ? 0 : 8;
  #pragma unroll
  for (int nt = 0; nt < 4; ++nt) {
    int   cg   = nt * 16 + (lane & 15);
    float bias = bs[cg];
    #pragma unroll
    for (int r = 0; r < 8; ++r) {
      float v    = acc[nt][r] + bias;
      int   srow = s0 + wave * 16 + mr_lo + r;
      if (ty == 2) {
        Vtw[((size_t)(b * N_HEADS + h) * HEAD + cg) * SEQ + srow] = f2bf(v);
      } else {
        bf16* dst = (ty == 0) ? Qw : Kw;
        dst[((size_t)(b * N_HEADS + h) * SEQ + srow) * HEAD + cg] = f2bf(v);
      }
    }
  }
}

// ---------------------------------------------------------------------------
// Kernel 2: flash attention.  grid = (qtiles=16, heads=16, batch=2), block=256.
// Each wave owns 16 query rows; block streams 64-key chunks through LDS
// using async global->LDS DMA.
// ---------------------------------------------------------------------------
__global__ __launch_bounds__(256)
void attention_kernel(const bf16* __restrict__ Qw, const bf16* __restrict__ Kw,
                      const bf16* __restrict__ Vtw, bf16* __restrict__ Ow)
{
  const int qt   = blockIdx.x;
  const int h    = blockIdx.y;
  const int b    = blockIdx.z;
  const int tid  = threadIdx.x;
  const int wave = tid >> 5, lane = tid & 31;

  __shared__ __align__(16) bf16 k_s[64][72];        // [key][kdim]
  __shared__ __align__(16) bf16 v_s[64][72];        // [kdim][key]  (from V^T)
  __shared__ __align__(16) bf16 p_s[8][16][72];     // per-wave P tile relayout

  const size_t bh = (size_t)(b * N_HEADS + h);
  const int    q0 = qt * 128 + wave * 16;
  const bf16* qptr = Qw + (bh * SEQ + q0) * HEAD;

  const int row  = lane & 15;
  const int klo  = (lane >> 4) * 8;
  const int col  = lane & 15;
  const int koff = (lane >> 4) * 16;

  // Q fragments for both 32-wide K-steps, resident in VGPRs for whole kernel
  FragAB aQ[2];
  #pragma unroll
  for (int s = 0; s < 2; ++s) {
    aQ[s].q[0] = *(const uint4*)&qptr[row * HEAD + s * 32 + klo];
    aQ[s].q[1] = *(const uint4*)&qptr[row * HEAD + s * 32 + 16 + klo];
  }

  v8f   oacc[4] = {};
  float mrow[8], lrow[8];
  #pragma unroll
  for (int r = 0; r < 8; ++r) { mrow[r] = -3.0e38f; lrow[r] = 0.f; }

  // per-thread staging coordinates: 64 rows x 64 cols, 32B per thread
  const int sr = tid >> 2;              // 0..63
  const int sc = (tid & 3) * 16;        // 0,16,32,48

  for (int c0 = 0; c0 < SEQ; c0 += 64) {
    __syncthreads();
    { // stage K chunk + V chunk via async DMA to LDS (no VGPR round-trip)
      unsigned           lk = lds_off(&k_s[sr][sc]);
      unsigned long long gk = (unsigned long long)(uintptr_t)
                              &Kw[(bh * SEQ + c0 + sr) * HEAD + sc];
      ASYNC_G2L_B128(lk, gk, 0);
      ASYNC_G2L_B128(lk, gk, 16);
      unsigned           lv = lds_off(&v_s[sr][sc]);
      unsigned long long gv = (unsigned long long)(uintptr_t)
                              &Vtw[(bh * HEAD + sr) * SEQ + c0 + sc];
      ASYNC_G2L_B128(lv, gv, 0);
      ASYNC_G2L_B128(lv, gv, 16);
    }
    asm volatile("s_wait_asynccnt 0" ::: "memory");
    __syncthreads();

    // S = Q K^T : 16 x 64, f32
    v8f sacc[4] = {};
    #pragma unroll
    for (int ks = 0; ks < 2; ++ks) {
      #pragma unroll
      for (int nt = 0; nt < 4; ++nt) {
        FragAB fb;
        fb.q[0] = *(const uint4*)&k_s[nt * 16 + col][ks * 32 + klo];
        fb.q[1] = *(const uint4*)&k_s[nt * 16 + col][ks * 32 + 16 + klo];
        sacc[nt] = WMMA_BF16(aQ[ks].v, fb.v, sacc[nt]);
      }
    }
    #pragma unroll
    for (int nt = 0; nt < 4; ++nt)
      #pragma unroll
      for (int r = 0; r < 8; ++r) sacc[nt][r] *= 0.125f;   // 1/sqrt(64)

    // online softmax: reductions stay inside 16-lane halves (rows 0-7 / 8-15)
    float alpha[8];
    #pragma unroll
    for (int r = 0; r < 8; ++r) {
      float mx = fmaxf(fmaxf(sacc[0][r], sacc[1][r]), fmaxf(sacc[2][r], sacc[3][r]));
      #pragma unroll
      for (int off = 1; off < 16; off <<= 1) mx = fmaxf(mx, __shfl_xor(mx, off, 32));
      float mnew = fmaxf(mrow[r], mx);
      alpha[r] = __expf(mrow[r] - mnew);
      mrow[r]  = mnew;
    }
    #pragma unroll
    for (int r = 0; r < 8; ++r) {
      float rs   = 0.f;
      int   prow = (lane < 16) ? r : 8 + r;
      #pragma unroll
      for (int nt = 0; nt < 4; ++nt) {
        float p = __expf(sacc[nt][r] - mrow[r]);
        rs += p;
        p_s[wave][prow][nt * 16 + col] = f2bf(p);   // C-layout -> A-layout via LDS
      }
      #pragma unroll
      for (int off = 1; off < 16; off <<= 1) rs += __shfl_xor(rs, off, 32);
      lrow[r] = lrow[r] * alpha[r] + rs;
      #pragma unroll
      for (int nt = 0; nt < 4; ++nt) oacc[nt][r] *= alpha[r];
    }
    __syncthreads();

    // O += P V
    #pragma unroll
    for (int ks = 0; ks < 2; ++ks) {
      FragAB fp;
      fp.q[0] = *(const uint4*)&p_s[wave][row][ks * 32 + klo];
      fp.q[1] = *(const uint4*)&p_s[wave][row][ks * 32 + 16 + klo];
      #pragma unroll
      for (int nt = 0; nt < 4; ++nt) {
        FragAB fv;
        fv.q[0] = *(const uint4*)&v_s[nt * 16 + col][ks * 32 + klo];
        fv.q[1] = *(const uint4*)&v_s[nt * 16 + col][ks * 32 + 16 + klo];
        oacc[nt] = WMMA_BF16(fp.v, fv.v, oacc[nt]);
      }
    }
  }

  // normalize, concat heads: O[token][h*64 + k], bf16
  const int mr_lo = (lane < 16) ? 0 : 8;
  #pragma unroll
  for (int nt = 0; nt < 4; ++nt) {
    #pragma unroll
    for (int r = 0; r < 8; ++r) {
      float v   = oacc[nt][r] / lrow[r];
      int   tok = b * SEQ + q0 + mr_lo + r;
      Ow[(size_t)tok * D_MODEL + h * HEAD + nt * 16 + col] = f2bf(v);
    }
  }
}

// ---------------------------------------------------------------------------
// Kernel 3: output projection.  grid = (Mtiles=32, Ntiles=16), block=256.
// out[m][e] = sum_d O[m][d] * Wo[e][d] + bo[e]   (f32 output)
// ---------------------------------------------------------------------------
__global__ __launch_bounds__(256)
void out_proj_kernel(const bf16* __restrict__ Ow, const float* __restrict__ Wo,
                     const float* __restrict__ bo, float* __restrict__ out)
{
  const int mt    = blockIdx.x;
  const int ntile = blockIdx.y;
  const int tid   = threadIdx.x;
  const int wave  = tid >> 5, lane = tid & 31;

  __shared__ __align__(16) bf16 a_s[128][40];
  __shared__ __align__(16) bf16 bT_s[64][40];   // [e][d] slice of Wo (already B^T-friendly)

  const int m0  = mt * 128;
  const int n0g = ntile * 64;
  v8f acc[4] = {};

  for (int k0 = 0; k0 < D_MODEL; k0 += 32) {
    __syncthreads();
    { // stage A from bf16 workspace
      int row = tid >> 1, c = (tid & 1) * 16;
      const uint4* src = (const uint4*)&Ow[(size_t)(m0 + row) * D_MODEL + k0 + c];
      uint4* dst = (uint4*)&a_s[row][c];
      dst[0] = src[0]; dst[1] = src[1];
    }
    { // stage B: bT_s[n][k] = Wo[(n0g+n)*1024 + k0+k] (contiguous b128 LDS store)
      int n = tid >> 2, kk = (tid & 3) * 8;
      const float4* src = (const float4*)&Wo[(size_t)(n0g + n) * D_MODEL + k0 + kk];
      float4 f0 = src[0], f1 = src[1];
      uint4 p = { pk2(f0.x,f0.y), pk2(f0.z,f0.w), pk2(f1.x,f1.y), pk2(f1.z,f1.w) };
      *(uint4*)&bT_s[n][kk] = p;
      if (k0 + 32 < D_MODEL)
        __builtin_prefetch(&Wo[(size_t)(n0g + n) * D_MODEL + k0 + 32 + kk], 0, 0);
    }
    __syncthreads();

    const int row  = lane & 15, klo  = (lane >> 4) * 8;
    const int col  = lane & 15, koff = (lane >> 4) * 16;
    FragAB fa;
    fa.q[0] = *(const uint4*)&a_s[wave * 16 + row][klo];
    fa.q[1] = *(const uint4*)&a_s[wave * 16 + row][16 + klo];
    #pragma unroll
    for (int nt = 0; nt < 4; ++nt) {
      FragAB fb;
      fb.q[0] = *(const uint4*)&bT_s[nt * 16 + col][klo];
      fb.q[1] = *(const uint4*)&bT_s[nt * 16 + col][16 + klo];
      acc[nt] = WMMA_BF16(fa.v, fb.v, acc[nt]);
    }
  }

  const int mr_lo = (lane < 16) ? 0 : 8;
  #pragma unroll
  for (int nt = 0; nt < 4; ++nt) {
    int   cg   = n0g + nt * 16 + (lane & 15);
    float bias = bo[cg];
    #pragma unroll
    for (int r = 0; r < 8; ++r)
      out[(size_t)(m0 + wave * 16 + mr_lo + r) * D_MODEL + cg] = acc[nt][r] + bias;
  }
}

// ---------------------------------------------------------------------------
extern "C" void kernel_launch(void* const* d_in, const int* in_sizes, int n_in,
                              void* d_out, int out_size, void* d_ws, size_t ws_size,
                              hipStream_t stream) {
  const float* x  = (const float*)d_in[0];
  const float* Wq = (const float*)d_in[1];
  const float* bq = (const float*)d_in[2];
  const float* Wk = (const float*)d_in[3];
  const float* bk = (const float*)d_in[4];
  const float* Wv = (const float*)d_in[5];
  const float* bv = (const float*)d_in[6];
  const float* Wo = (const float*)d_in[7];
  const float* bo = (const float*)d_in[8];
  float* out = (float*)d_out;

  // Workspace layout (bf16): Q | K | V^T | O  -> 4 * 8 MB = 32 MB (fits in L2)
  const size_t QKV = (size_t)BATCH * N_HEADS * SEQ * HEAD;   // 4M elements
  bf16* Qw  = (bf16*)d_ws;
  bf16* Kw  = Qw + QKV;
  bf16* Vtw = Kw + QKV;
  bf16* Ow  = Vtw + QKV;

  qkv_proj_kernel<<<dim3(TOKENS / 128, N_HEADS, 3), 256, 0, stream>>>(
      x, Wq, bq, Wk, bk, Wv, bv, Qw, Kw, Vtw);
  attention_kernel<<<dim3(SEQ / 128, N_HEADS, BATCH), 256, 0, stream>>>(
      Qw, Kw, Vtw, Ow);
  out_proj_kernel<<<dim3(TOKENS / 128, D_MODEL / 64), 256, 0, stream>>>(
      Ow, Wo, bo, out);
}
